// NCA_grow_magnitude_24953759989918
// MI455X (gfx1250) — matrix operand order repaired and checked by reading, not compiled, not run on hardware
//
#include <hip/hip_runtime.h>
#include <hip/hip_bf16.h>
#include <float.h>
#include <math.h>

typedef _Float16 v16h __attribute__((ext_vector_type(16)));
typedef _Float16 v8h  __attribute__((ext_vector_type(8)));
typedef float    v8f  __attribute__((ext_vector_type(8)));
typedef unsigned int u32x4 __attribute__((ext_vector_type(4)));
typedef int          i32x8 __attribute__((ext_vector_type(8)));
typedef int          i32x4 __attribute__((ext_vector_type(4)));

#if defined(__has_builtin)
#  if __has_builtin(__builtin_amdgcn_tensor_load_to_lds) && \
      __has_builtin(__builtin_amdgcn_s_wait_tensorcnt)
#    define HAVE_TDM 1
#  endif
#endif
#ifndef HAVE_TDM
#  define HAVE_TDM 0
#endif

#define B_   32
#define C_   16
#define H_   256
#define W_   256
#define HID_ 128
#define HW_  (H_*W_)

// ---------------- LDS layout for the main fused kernel ----------------
#define XS_N      (C_*20*20)                 // x tile, halo 2      : 6400 f32
#define MASK_N    (18*20)                    // fire mask, halo 1   : 360 f32
#define CONST_N   64                         // norm constants      : 64 f32
#define B1_N      128                        // bias                : 128 f32
#define XNA_N     (18*20)                    // x_new alpha, halo 1 : 360 f32
#define XNEW_N    (256*17)                   // x_new inner (pad 17): 4352 f32
#define FLOATS_N  (XS_N+MASK_N+CONST_N+B1_N+XNA_N+XNEW_N)  // 11664 floats
#define YF_N      (336*72)                   // y features f16 (row pad 72)
#define W1S_N     (128*72)                   // w1 f16 (row pad 72)
#define W2S_N     (16*136)                   // w2 f16 (row pad 136)
#define HB_N      (8*16*136)                 // per-wave h staging
#define SMEM_BYTES (FLOATS_N*4 + (YF_N+W1S_N+W2S_N+HB_N)*2)   // 152640 B

__device__ __forceinline__ v16h cat8(v8h lo, v8h hi) {
  v16h r;
#pragma unroll
  for (int i = 0; i < 8; ++i) { r[i] = lo[i]; r[i + 8] = hi[i]; }
  return r;
}
__device__ __forceinline__ v8h ld8(const _Float16* p) {
  return *reinterpret_cast<const v8h*>(p);
}

// ---------------- Kernel 1: convert weights to f16 (K-pad w1 to 64) ----
__global__ __launch_bounds__(256) void k_convert_weights(
    const float* __restrict__ w1, const float* __restrict__ w2,
    _Float16* __restrict__ w1h, _Float16* __restrict__ w2h) {
  int idx = blockIdx.x * 256 + threadIdx.x;
  if (idx < HID_ * 64) {
    int o = idx >> 6, k = idx & 63;
    w1h[idx] = (k < 48) ? (_Float16)w1[o * 48 + k] : (_Float16)0.f;
  } else if (idx < HID_ * 64 + C_ * HID_) {
    int i2 = idx - HID_ * 64;
    w2h[i2] = (_Float16)w2[i2];
  }
}

// ---------------- Kernel 2: per-channel Gx/Gy min/max partials ---------
__global__ __launch_bounds__(256) void k_minmax_partial(
    const float* __restrict__ x, float* __restrict__ partials) {
  float mnx[C_], mxx[C_], mny[C_], mxy[C_];
#pragma unroll
  for (int c = 0; c < C_; ++c) {
    mnx[c] = FLT_MAX; mxx[c] = -FLT_MAX;
    mny[c] = FLT_MAX; mxy[c] = -FLT_MAX;
  }
  const int tid = threadIdx.x;
  const int stride = 256 * 256;
  for (int pid = blockIdx.x * 256 + tid; pid < B_ * HW_; pid += stride) {
    int b = pid >> 16;
    int rem = pid & 65535;
    int h = rem >> 8, w = rem & 255;
    bool hm = h > 0, hp = h < H_ - 1, wm = w > 0, wp = w < W_ - 1;
    const float* xb = x + (size_t)b * C_ * HW_ + h * W_ + w;
#pragma unroll
    for (int c = 0; c < C_; ++c) {
      const float* xc = xb + c * HW_;
      float a00 = (hm && wm) ? xc[-W_ - 1] : 0.f;
      float a01 = hm ? xc[-W_] : 0.f;
      float a02 = (hm && wp) ? xc[-W_ + 1] : 0.f;
      float a10 = wm ? xc[-1] : 0.f;
      float a12 = wp ? xc[1] : 0.f;
      float a20 = (hp && wm) ? xc[W_ - 1] : 0.f;
      float a21 = hp ? xc[W_] : 0.f;
      float a22 = (hp && wp) ? xc[W_ + 1] : 0.f;
      float gx = ((a02 - a00) + 2.f * (a12 - a10) + (a22 - a20)) * 0.125f;
      float gy = ((a20 - a00) + 2.f * (a21 - a01) + (a22 - a02)) * 0.125f;
      mnx[c] = fminf(mnx[c], gx); mxx[c] = fmaxf(mxx[c], gx);
      mny[c] = fminf(mny[c], gy); mxy[c] = fmaxf(mxy[c], gy);
    }
  }
  // wave32 shuffle reduce
#pragma unroll
  for (int c = 0; c < C_; ++c) {
#pragma unroll
    for (int off = 16; off > 0; off >>= 1) {
      mnx[c] = fminf(mnx[c], __shfl_down(mnx[c], off));
      mxx[c] = fmaxf(mxx[c], __shfl_down(mxx[c], off));
      mny[c] = fminf(mny[c], __shfl_down(mny[c], off));
      mxy[c] = fmaxf(mxy[c], __shfl_down(mxy[c], off));
    }
  }
  __shared__ float warr[8][64];
  int lane = tid & 31, wv = tid >> 5;
  if (lane == 0) {
#pragma unroll
    for (int c = 0; c < C_; ++c) {
      warr[wv][c]      = mnx[c];
      warr[wv][16 + c] = mxx[c];
      warr[wv][32 + c] = mny[c];
      warr[wv][48 + c] = mxy[c];
    }
  }
  __syncthreads();
  if (tid < 64) {
    int kind = tid >> 4;
    bool isMin = (kind == 0) || (kind == 2);
    float v = warr[0][tid];
#pragma unroll
    for (int wi = 1; wi < 8; ++wi)
      v = isMin ? fminf(v, warr[wi][tid]) : fmaxf(v, warr[wi][tid]);
    partials[blockIdx.x * 64 + tid] = v;
  }
}

// ---------------- Kernel 3: finalize norm constants --------------------
__global__ __launch_bounds__(64) void k_minmax_final(
    const float* __restrict__ partials, float* __restrict__ consts) {
  __shared__ float red[64];
  int q = threadIdx.x;
  int kind = q >> 4;
  bool isMin = (kind == 0) || (kind == 2);
  float v = isMin ? FLT_MAX : -FLT_MAX;
  for (int b = 0; b < 256; ++b) {
    float pv = partials[b * 64 + q];
    v = isMin ? fminf(v, pv) : fmaxf(v, pv);
  }
  red[q] = v;
  __syncthreads();
  if (q < 16) {
    float mna = fabsf(red[q]);        // |min Gx|
    float mxa = fabsf(red[16 + q]);   // |max Gx|
    float dvx = mna + mxa;
    consts[q]      = mna;
    consts[16 + q] = (dvx == 0.f) ? 1.f : (1.f / dvx);
    float mnb = fabsf(red[32 + q]);   // |min Gy|
    float mxb = fabsf(red[48 + q]);   // |max Gy|
    float dvy = mnb + mxb;
    consts[32 + q] = mnb;
    consts[48 + q] = (dvy == 0.f) ? 1.f : (1.f / dvy);
  }
}

// ---------------- Kernel 4: fused NCA step (WMMA + TDM) ----------------
__global__ __launch_bounds__(256) void k_nca_main(
    const float* __restrict__ x, const float* __restrict__ b1,
    const float* __restrict__ mask_rand,
    const _Float16* __restrict__ w1h, const _Float16* __restrict__ w2h,
    const float* __restrict__ consts, float* __restrict__ out) {
  extern __shared__ __align__(16) char smem_raw[];
  float* xs    = (float*)smem_raw;        // [16][20][20]
  float* maskS = xs + XS_N;               // [18][20]
  float* cS    = maskS + MASK_N;          // [64]
  float* b1S   = cS + CONST_N;            // [128]
  float* xna   = b1S + B1_N;              // [18][20]
  float* xnew  = xna + XNA_N;             // [256][17]
  _Float16* yf  = (_Float16*)(xnew + XNEW_N); // [336][72]
  _Float16* w1s = yf + YF_N;                  // [128][72]
  _Float16* w2s = w1s + W1S_N;                // [16][136]
  _Float16* hb  = w2s + W2S_N;                // [8 waves][16][136]

  const int tid = threadIdx.x;
  const int blk = blockIdx.x;
  const int b   = blk >> 8;
  const int t   = blk & 255;
  const int ty0 = (t >> 4) << 4;
  const int tx0 = (t & 15) << 4;
  const float* xb = x + (size_t)b * C_ * HW_;
  const int lane = tid & 31;
  const int wv   = tid >> 5;

  // ---- Phase 1: stage x/mask tiles, constants, weights into LDS ----
  // Interior tiles (no halo clamping needed) pull the 16x20x20 f32 x-tile
  // through the Tensor Data Mover, overlapping with weight staging below.
#if HAVE_TDM
  const bool tdm_tile = (ty0 >= 16 && ty0 <= 224 && tx0 >= 16 && tx0 <= 224);
  if (tdm_tile) {
    if (wv == 0) {
      unsigned long long gaddr =
          (unsigned long long)(const void*)(xb + (ty0 - 2) * W_ + (tx0 - 2));
      unsigned int laddr = (unsigned int)(unsigned long long)(void*)xs;
      // D# group 0: count=1, lds_addr, 57-bit global_addr, type=2 ("image")
      u32x4 g0 = { 1u,
                   laddr,
                   (unsigned int)(gaddr & 0xFFFFFFFFu),
                   (unsigned int)((gaddr >> 32) & 0x01FFFFFFu) | (2u << 30) };
      // D# group 1: data_size=4B, tile 20x20x16, dim0/dim1 strides 256/65536
      const unsigned int TD0 = 1u << 20, TD1 = 1u << 20;  // no OOB clamp wanted
      i32x8 g1 = { (int)(2u << 16),
                   (int)((TD0 & 0xFFFFu) << 16),
                   (int)((TD0 >> 16) | ((TD1 & 0xFFFFu) << 16)),
                   (int)((TD1 >> 16) | (20u << 16)),          // tile_dim0=20
                   (int)(20u | (16u << 16)),                  // tile_dim1=20, tile_dim2=16
                   (int)W_,                                   // dim0 stride = 256
                   (int)(((unsigned)HW_ & 0xFFFFu) << 16),    // dim1 stride lo16 (=0)
                   (int)((unsigned)HW_ >> 16) };              // dim1 stride hi (=1)
      // D# group 2: tensor_dim2=16, no 4th dim
      i32x4 g2 = { 16, 0, 0, 0 };
      i32x4 g3 = { 0, 0, 0, 0 };
      i32x8 g4 = { 0, 0, 0, 0, 0, 0, 0, 0 };  // unused VADDR4 slot (NULL)
      __builtin_amdgcn_tensor_load_to_lds(g0, g1, g2, g3, g4, 0);
    }
  } else
#endif
  {
    for (int i = tid; i < XS_N; i += 256) {
      int c = i / 400, pos = i % 400;
      int yy = pos / 20, xx = pos % 20;
      int gy = ty0 + yy - 2, gx = tx0 + xx - 2;
      float v = 0.f;
      if (gy >= 0 && gy < H_ && gx >= 0 && gx < W_) v = xb[c * HW_ + gy * W_ + gx];
      xs[i] = v;
    }
  }
  for (int i = tid; i < 324; i += 256) {
    int py = i / 18, px = i % 18;
    int gy = ty0 + py - 1, gx = tx0 + px - 1;
    float v = 1.f;  // fire=0 for OOB (unused anyway)
    if (gy >= 0 && gy < H_ && gx >= 0 && gx < W_)
      v = mask_rand[(size_t)b * HW_ + gy * W_ + gx];
    maskS[py * 20 + px] = v;
  }
  if (tid < CONST_N) cS[tid] = consts[tid];
  if (tid < B1_N)    b1S[tid] = b1[tid];
  for (int i = tid; i < HID_ * 64; i += 256) {
    int o = i >> 6, k = i & 63;
    w1s[o * 72 + k] = w1h[i];
  }
  for (int i = tid; i < C_ * HID_; i += 256) {
    int n = i >> 7, k = i & 127;
    w2s[n * 136 + k] = w2h[i];
  }
#if HAVE_TDM
  if (tdm_tile && wv == 0) __builtin_amdgcn_s_wait_tensorcnt(0);
#endif
  __syncthreads();

  // ---- Phase 2: perception -> y features (f16), A-fragment friendly ----
  for (int p = tid; p < 336; p += 256) {
    _Float16 feat[64];
    if (p < 324) {
      int py = p / 18, px = p % 18;
      const float* base0 = xs + py * 20 + px;
#pragma unroll
      for (int c = 0; c < C_; ++c) {
        const float* bp = base0 + c * 400;
        float a00 = bp[0],  a01 = bp[1],  a02 = bp[2];
        float a10 = bp[20], a11 = bp[21], a12 = bp[22];
        float a20 = bp[40], a21 = bp[41], a22 = bp[42];
        float gx = ((a02 - a00) + 2.f * (a12 - a10) + (a22 - a20)) * 0.125f;
        float gy = ((a20 - a00) + 2.f * (a21 - a01) + (a22 - a02)) * 0.125f;
        float gl = (a01 + a10 + a12 + a21 - 4.f * a11) * 0.125f;
        float gxn = (gx + cS[c])      * cS[16 + c];
        float gyn = (gy + cS[32 + c]) * cS[48 + c];
        float mag = sqrtf(gxn * gxn + gyn * gyn);
        feat[c]      = (_Float16)a11;
        feat[16 + c] = (_Float16)gl;
        feat[32 + c] = (_Float16)mag;
        feat[48 + c] = (_Float16)0.f;
      }
    } else {
#pragma unroll
      for (int i = 0; i < 64; ++i) feat[i] = (_Float16)0.f;
    }
    _Float16* dst = yf + p * 72;
#pragma unroll
    for (int j = 0; j < 8; ++j) {
      v8h v;
#pragma unroll
      for (int e = 0; e < 8; ++e) v[e] = feat[j * 8 + e];
      *reinterpret_cast<v8h*>(dst + j * 8) = v;
    }
  }
  __syncthreads();

  // ---- Phase 3: WMMA GEMMs per 16-pixel M-tile ----
  const int g  = lane >> 4;   // half-wave group
  const int mr = lane & 15;   // row (A/M) and column (B/N) index
  _Float16* hrow0 = hb + wv * (16 * 136);

  for (int mt = wv; mt < 21; mt += 8) {
    const _Float16* arow = yf + (mt * 16 + mr) * 72;
    v16h a0 = cat8(ld8(arow + 8 * g),      ld8(arow + 16 + 8 * g));
    v16h a1 = cat8(ld8(arow + 32 + 8 * g), ld8(arow + 48 + 8 * g));
#pragma unroll
    for (int nt = 0; nt < 8; ++nt) {
      const _Float16* brow = w1s + (nt * 16 + mr) * 72;
      v16h b0  = cat8(ld8(brow + 8 * g),      ld8(brow + 16 + 8 * g));
      v16h b1f = cat8(ld8(brow + 32 + 8 * g), ld8(brow + 48 + 8 * g));
      v8f acc = {};
      acc = __builtin_amdgcn_wmma_f32_16x16x32_f16(false, a0, false, b0,
                                                   (short)0, acc, false, false);
      acc = __builtin_amdgcn_wmma_f32_16x16x32_f16(false, a1, false, b1f,
                                                   (short)0, acc, false, false);
      float bias = b1S[nt * 16 + mr];
#pragma unroll
      for (int r = 0; r < 8; ++r) {
        float hv = acc[r] + bias;
        hv = hv > 0.f ? hv : 0.f;
        hrow0[(r + 8 * g) * 136 + nt * 16 + mr] = (_Float16)hv;
      }
    }
    __builtin_amdgcn_wave_barrier();  // per-wave LDS staging, in-order DS
    v8f acc2 = {};
#pragma unroll
    for (int kt = 0; kt < 4; ++kt) {
      const _Float16* hp = hrow0 + mr * 136 + kt * 32 + 8 * g;
      v16h ha = cat8(ld8(hp), ld8(hp + 16));
      const _Float16* wp = w2s + mr * 136 + kt * 32 + 8 * g;
      v16h wb = cat8(ld8(wp), ld8(wp + 16));
      acc2 = __builtin_amdgcn_wmma_f32_16x16x32_f16(false, ha, false, wb,
                                                    (short)0, acc2, false, false);
    }
#pragma unroll
    for (int r = 0; r < 8; ++r) {
      int m = r + 8 * g;
      int p = mt * 16 + m;
      if (p < 324) {
        int py = p / 18, px = p % 18;
        int gy = ty0 + py - 1, gx = tx0 + px - 1;
        bool inImg = (gy >= 0 && gy < H_ && gx >= 0 && gx < W_);
        float fire = (maskS[py * 20 + px] <= 0.5f) ? 1.f : 0.f;
        float xv = xs[mr * 400 + (py + 1) * 20 + (px + 1)];
        float xn = xv + acc2[r] * fire;
        if (mr == 3) xna[py * 20 + px] = inImg ? xn : 0.f;
        if (py >= 1 && py < 17 && px >= 1 && px < 17)
          xnew[((py - 1) * 16 + (px - 1)) * 17 + mr] = xn;
      }
    }
  }
  __syncthreads();

  // ---- Phase 4: alive masks + masked writeout ----
  {
    int iy = tid >> 4, ix = tid & 15;
    float mpre = -FLT_MAX, mpost = -FLT_MAX;
#pragma unroll
    for (int dy = 0; dy < 3; ++dy)
#pragma unroll
      for (int dx = 0; dx < 3; ++dx) {
        mpre  = fmaxf(mpre,  xs[3 * 400 + (iy + 1 + dy) * 20 + (ix + 1 + dx)]);
        mpost = fmaxf(mpost, xna[(iy + dy) * 20 + (ix + dx)]);
      }
    float life = (mpre > 0.1f && mpost > 0.1f) ? 1.f : 0.f;
    int gy = ty0 + iy, gx = tx0 + ix;
    float* ob = out + (size_t)b * C_ * HW_ + gy * W_ + gx;
#pragma unroll
    for (int c = 0; c < C_; ++c)
      ob[c * HW_] = xnew[tid * 17 + c] * life;
  }
}

// ---------------- Launch ----------------------------------------------
extern "C" void kernel_launch(void* const* d_in, const int* in_sizes, int n_in,
                              void* d_out, int out_size, void* d_ws, size_t ws_size,
                              hipStream_t stream) {
  (void)in_sizes; (void)n_in; (void)out_size; (void)ws_size;
  const float* x  = (const float*)d_in[0];
  const float* w1 = (const float*)d_in[1];
  const float* b1 = (const float*)d_in[2];
  const float* w2 = (const float*)d_in[3];
  const float* mr = (const float*)d_in[4];
  float* out = (float*)d_out;

  float* partials = (float*)d_ws;              // 256*64 f32
  float* consts   = partials + 256 * 64;       // 64 f32
  _Float16* w1h   = (_Float16*)(consts + 64);  // 128*64 f16
  _Float16* w2h   = w1h + HID_ * 64;           // 16*128 f16

  k_convert_weights<<<40, 256, 0, stream>>>(w1, w2, w1h, w2h);
  k_minmax_partial<<<256, 256, 0, stream>>>(x, partials);
  k_minmax_final<<<1, 64, 0, stream>>>(partials, consts);
  k_nca_main<<<B_ * 256, 256, SMEM_BYTES, stream>>>(x, b1, mr, w1h, w2h,
                                                    consts, out);
}
